// ProbSparseSelfAttention_42253888258545
// MI455X (gfx1250) — matrix operand order repaired and checked by reading, not compiled, not run on hardware
//
#include <hip/hip_runtime.h>
#include <hip/hip_bf16.h>

// ---------------------------------------------------------------------------
// MHA forward, B=4, S=2048, D=1024, H=16, depth=64, gfx1250 (CDNA5).
// All matmuls: V_WMMA_F32_16X16X32_BF16 (bf16 in, f32 accumulate).
// Attention K tile staged by the Tensor Data Mover (tensor_load_to_lds +
// s_wait_tensorcnt); softmax row reductions via DPP16 butterflies.
// ---------------------------------------------------------------------------

typedef __attribute__((ext_vector_type(16))) __bf16 bf16x16;
typedef __attribute__((ext_vector_type(8)))  float  f32x8;
typedef __attribute__((ext_vector_type(4)))  unsigned int u32x4;
typedef __attribute__((ext_vector_type(8)))  int    i32x8;
typedef __attribute__((ext_vector_type(4)))  int    i32x4;

union FragU { uint4 u[2]; bf16x16 v; };
union U8bf  { uint4 u; __bf16 b[8]; };

__device__ __forceinline__ bf16x16 ld_frag(const __bf16* p0, const __bf16* p1) {
  FragU f;
  f.u[0] = *(const uint4*)p0;   // frag elements 0..7
  f.u[1] = *(const uint4*)p1;   // frag elements 8..15
  return f.v;
}

__device__ __forceinline__ f32x8 wmma_bf16(bf16x16 a, bf16x16 b, f32x8 c) {
  return __builtin_amdgcn_wmma_f32_16x16x32_bf16(
      false, a, false, b, (short)0, c, false, false);
}

// --- DPP16 butterfly reductions across one 16-lane row (stays in half-wave) --
template <int CTRL>
__device__ __forceinline__ float dpp_xf(float x) {
  int xi = __float_as_int(x);
#if __has_builtin(__builtin_amdgcn_mov_dpp)
  int r = __builtin_amdgcn_mov_dpp(xi, CTRL, 0xF, 0xF, false);
#else
  int r = __builtin_amdgcn_update_dpp(xi, xi, CTRL, 0xF, 0xF, false);
#endif
  return __int_as_float(r);
}
__device__ __forceinline__ float row16_max(float x) {
  x = fmaxf(x, dpp_xf<0xB1>(x));    // quad_perm(1,0,3,2)
  x = fmaxf(x, dpp_xf<0x4E>(x));    // quad_perm(2,3,0,1)
  x = fmaxf(x, dpp_xf<0x141>(x));   // row_half_mirror
  x = fmaxf(x, dpp_xf<0x140>(x));   // row_mirror
  return x;
}
__device__ __forceinline__ float row16_sum(float x) {
  x += dpp_xf<0xB1>(x);
  x += dpp_xf<0x4E>(x);
  x += dpp_xf<0x141>(x);
  x += dpp_xf<0x140>(x);
  return x;
}

// ---------------------------------------------------------------------------
// GEMM: Out[M,N] = (X[M,K] @ W[K,N] + bias[N]) * oscale
// 64x128 workgroup tile, BK=32, 256 threads (8 waves).
// Wave w: m-tile (w&3), n-tiles (w>>2)*4 .. +3  => 4 WMMA / K-step, A reused 4x.
// ---------------------------------------------------------------------------
template <typename TIn, bool OUT_F32>
__global__ __launch_bounds__(256)
void gemm_bias_kernel(const TIn* __restrict__ X, const float* __restrict__ W,
                      const float* __restrict__ bias, void* __restrict__ Out,
                      int M, int K, int N, float oscale) {
  __shared__ __align__(16) __bf16 As[64][40];     // [m][k]
  __shared__ __align__(16) __bf16 BsT[128][40];   // [n][k] (transposed)

  const int tid  = threadIdx.x;
  const int lane = tid & 31;
  const int wave = tid >> 5;
  const int m0 = blockIdx.y * 64;
  const int n0 = blockIdx.x * 128;

  const int mt = wave & 3;
  const int nq = wave >> 2;

  const int lm = lane & 15;
  const int hi = lane >> 4;
  const int ah = hi * 8;
  const int kh = hi * 16;

  f32x8 acc[4] = {};

  for (int k0 = 0; k0 < K; k0 += 32) {
    {
      const int row = tid >> 2;
      const int cb  = (tid & 3) * 8;
      const TIn* src = X + (size_t)(m0 + row) * K + k0 + cb;
      if constexpr (sizeof(TIn) == 4) {
        float4 f0 = *(const float4*)(src);
        float4 f1 = *(const float4*)(src + 4);
        __bf16* d = &As[row][cb];
        d[0] = (__bf16)f0.x; d[1] = (__bf16)f0.y;
        d[2] = (__bf16)f0.z; d[3] = (__bf16)f0.w;
        d[4] = (__bf16)f1.x; d[5] = (__bf16)f1.y;
        d[6] = (__bf16)f1.z; d[7] = (__bf16)f1.w;
      } else {
        *(uint4*)&As[row][cb] = *(const uint4*)src;
      }
    }
    {
      const int kr = tid >> 3;          // 0..31
      const int cb = (tid & 7) * 16;    // 0..112
      const float* src = W + (size_t)(k0 + kr) * N + n0 + cb;
#pragma unroll
      for (int g = 0; g < 2; ++g) {
        float4 f0 = *(const float4*)(src + g * 8);
        float4 f1 = *(const float4*)(src + g * 8 + 4);
        const int c = cb + g * 8;
        BsT[c + 0][kr] = (__bf16)f0.x; BsT[c + 1][kr] = (__bf16)f0.y;
        BsT[c + 2][kr] = (__bf16)f0.z; BsT[c + 3][kr] = (__bf16)f0.w;
        BsT[c + 4][kr] = (__bf16)f1.x; BsT[c + 5][kr] = (__bf16)f1.y;
        BsT[c + 6][kr] = (__bf16)f1.z; BsT[c + 7][kr] = (__bf16)f1.w;
      }
    }
    __syncthreads();

    bf16x16 a = ld_frag(&As[mt * 16 + lm][ah], &As[mt * 16 + lm][16 + ah]);
#pragma unroll
    for (int j = 0; j < 4; ++j) {
      const int nt = (nq * 4 + j) * 16;
      bf16x16 b = ld_frag(&BsT[nt + lm][ah], &BsT[nt + lm][16 + ah]);
      acc[j] = wmma_bf16(a, b, acc[j]);
    }
    __syncthreads();
  }

#pragma unroll
  for (int j = 0; j < 4; ++j) {
    const int colg = n0 + (nq * 4 + j) * 16 + lm;
    const float bb = bias[colg];
#pragma unroll
    for (int v = 0; v < 8; ++v) {
      const int rowg = m0 + mt * 16 + hi * 8 + v;
      const float r = (acc[j][v] + bb) * oscale;
      if constexpr (OUT_F32) ((float*)Out)[(size_t)rowg * N + colg] = r;
      else                   ((__bf16*)Out)[(size_t)rowg * N + colg] = (__bf16)r;
    }
  }
}

// ---------------------------------------------------------------------------
// Flash attention: workgroup = 128 q rows of one (b,h); wave = 16 q rows.
// kv tile = 64 (4 logit sub-tiles, 16 WMMA per step, 32 steps over S=2048).
// Q pre-scaled by (1/sqrt(64))*log2(e) at projection time -> softmax in 2^x.
// ---------------------------------------------------------------------------
__global__ __launch_bounds__(256)
void attn_kernel(const __bf16* __restrict__ Q, const __bf16* __restrict__ Km,
                 const __bf16* __restrict__ Vm, __bf16* __restrict__ O, int S) {
  __shared__ __align__(16) __bf16 Kt[64][72];       // [kv][depth], 128B row + 16B pad
  __shared__ __align__(16) __bf16 VtT[64][72];      // [depth][kv]
  __shared__ __align__(16) __bf16 Pst[8][16][72];   // per-wave P staging [q][kv]

  const int tid = threadIdx.x, lane = tid & 31, wave = tid >> 5;
  const int b = blockIdx.y >> 4, h = blockIdx.y & 15;
  const int lm = lane & 15, hi = lane >> 4, ah = hi * 8, kh = hi * 16;
  const size_t base = ((size_t)b * S) * 1024 + (size_t)h * 64;

  const int q0 = blockIdx.x * 128 + wave * 16;
  const unsigned kt_lds = (unsigned)(size_t)&Kt[0][0];

  // Q A-frags (depth chunks 0..31, 32..63) resident for the whole pass
  const __bf16* qp = Q + base + (size_t)(q0 + lm) * 1024;
  const bf16x16 aq0 = ld_frag(qp + ah,      qp + 16 + ah);
  const bf16x16 aq1 = ld_frag(qp + 32 + ah, qp + 48 + ah);

  float mrow[8], lrow[8];
  f32x8 oacc[4] = {};
#pragma unroll
  for (int v = 0; v < 8; ++v) { mrow[v] = -3.0e38f; lrow[v] = 0.0f; }

  for (int kv0 = 0; kv0 < S; kv0 += 64) {
    __syncthreads();
    // --- K tile via Tensor Data Mover: 2D tile 64x64 bf16, padded rows ------
    if (wave == 0) {
      const unsigned long long ga =
          (unsigned long long)(size_t)(Km + base + (size_t)kv0 * 1024);
      u32x4 g0;
      g0.x = 1u;                                       // count=1, user mode
      g0.y = kt_lds;                                   // lds_addr (bytes)
      g0.z = (unsigned)ga;                             // global_addr[31:0]
      g0.w = (unsigned)((ga >> 32) & 0x01FFFFFFu) | (2u << 30); // addr | type=2
      i32x8 g1;
      g1[0] = (1 << 16) | (1 << 20) | (4 << 22) | (3 << 25);
      //       data=2B    pad_en      intv=32dw   amount=4dw  (-> 72-bf16 rows)
      g1[1] = (int)(1024u << 16);   // tensor_dim0 = 1024
      g1[2] = (int)(2048u << 16);   // tensor_dim1 = 2048
      g1[3] = (int)(64u << 16);     // tile_dim0 = 64
      g1[4] = 64;                   // tile_dim1 = 64
      g1[5] = 1024;                 // tensor_dim0_stride = 1024 elements
      g1[6] = 0; g1[7] = 0;
      i32x4 z4 = {};
#if defined(__clang_major__) && (__clang_major__ >= 23)
      i32x8 z8 = {};
      __builtin_amdgcn_tensor_load_to_lds(g0, g1, z4, z4, z8, 0);
#else
      __builtin_amdgcn_tensor_load_to_lds(g0, g1, z4, z4, 0);
#endif
      __builtin_amdgcn_s_wait_tensorcnt(0);
    }
    // --- V tile 64x64: cooperative load + transpose into VtT ----------------
    {
      const int r  = tid >> 2;           // 0..63 kv row
      const int cb = (tid & 3) * 16;     // 0..48 depth col base
#pragma unroll
      for (int g = 0; g < 2; ++g) {
        const size_t goff = base + (size_t)(kv0 + r) * 1024 + cb + g * 8;
        U8bf vd; vd.u = *(const uint4*)(Vm + goff);
#pragma unroll
        for (int i = 0; i < 8; ++i) VtT[cb + g * 8 + i][r] = vd.b[i];
      }
      if (kv0 + 64 < S) {
        const size_t gnext = base + (size_t)(kv0 + 64 + r) * 1024 + cb;
        __builtin_prefetch(Vm + gnext, 0, 0);
        __builtin_prefetch(Km + gnext, 0, 0);
      }
    }
    __syncthreads();

    // logits: 4 kv sub-tiles, K-dim = depth 64 (2 chunks) -> 8 WMMA
    f32x8 s[4] = {{}, {}, {}, {}};
#pragma unroll
    for (int kc = 0; kc < 2; ++kc) {
      const bf16x16 aq = kc ? aq1 : aq0;
#pragma unroll
      for (int j = 0; j < 4; ++j) {
        bf16x16 bk = ld_frag(&Kt[j * 16 + lm][kc * 32 + ah],
                             &Kt[j * 16 + lm][kc * 32 + 16 + ah]);
        s[j] = wmma_bf16(aq, bk, s[j]);
      }
    }

    // online softmax over 64 columns; rows = vgpr v (+8*hi)
#pragma unroll
    for (int v = 0; v < 8; ++v) {
      const float t0 = s[0][v], t1 = s[1][v], t2 = s[2][v], t3 = s[3][v];
      const float rmax = row16_max(fmaxf(fmaxf(t0, t1), fmaxf(t2, t3)));
      const float mnew = fmaxf(mrow[v], rmax);
      const float corr = exp2f(mrow[v] - mnew);
      mrow[v] = mnew;
      const float p0 = exp2f(t0 - mnew);
      const float p1 = exp2f(t1 - mnew);
      const float p2 = exp2f(t2 - mnew);
      const float p3 = exp2f(t3 - mnew);
      lrow[v] = lrow[v] * corr + row16_sum((p0 + p1) + (p2 + p3));
#pragma unroll
      for (int d = 0; d < 4; ++d) oacc[d][v] *= corr;
      __bf16* pr = &Pst[wave][hi * 8 + v][lm];
      pr[0]  = (__bf16)p0;
      pr[16] = (__bf16)p1;
      pr[32] = (__bf16)p2;
      pr[48] = (__bf16)p3;
    }

    // P·V: two A-frags (kv 0..31, 32..63), 8 WMMA
    const bf16x16 ap0 = ld_frag(&Pst[wave][lm][ah],      &Pst[wave][lm][16 + ah]);
    const bf16x16 ap1 = ld_frag(&Pst[wave][lm][32 + ah], &Pst[wave][lm][48 + ah]);
#pragma unroll
    for (int d = 0; d < 4; ++d) {
      bf16x16 bv0 = ld_frag(&VtT[d * 16 + lm][ah],      &VtT[d * 16 + lm][16 + ah]);
      bf16x16 bv1 = ld_frag(&VtT[d * 16 + lm][32 + ah], &VtT[d * 16 + lm][48 + ah]);
      oacc[d] = wmma_bf16(ap0, bv0, oacc[d]);
      oacc[d] = wmma_bf16(ap1, bv1, oacc[d]);
    }
  }

  // normalize and store
#pragma unroll
  for (int v = 0; v < 8; ++v) {
    const float inv = 1.0f / lrow[v];
    const size_t orow = base + (size_t)(q0 + hi * 8 + v) * 1024;
#pragma unroll
    for (int d = 0; d < 4; ++d)
      O[orow + d * 16 + lm] = (__bf16)(oacc[d][v] * inv);
  }
}

// ---------------------------------------------------------------------------
extern "C" void kernel_launch(void* const* d_in, const int* in_sizes, int n_in,
                              void* d_out, int out_size, void* d_ws, size_t ws_size,
                              hipStream_t stream) {
  const float* v  = (const float*)d_in[0];
  const float* k  = (const float*)d_in[1];
  const float* q  = (const float*)d_in[2];
  const float* wq = (const float*)d_in[3];
  const float* bq = (const float*)d_in[4];
  const float* wk = (const float*)d_in[5];
  const float* bk = (const float*)d_in[6];
  const float* wv = (const float*)d_in[7];
  const float* bv = (const float*)d_in[8];
  const float* wo = (const float*)d_in[9];
  const float* bo = (const float*)d_in[10];

  const int B = 4, S = 2048, D = 1024, H = 16;
  const int M = B * S;
  const size_t elems = (size_t)M * D;

  __bf16* Qb = (__bf16*)d_ws;     // bf16 [B,S,1024] each
  __bf16* Kb = Qb + elems;
  __bf16* Vb = Kb + elems;
  __bf16* Ob = Vb + elems;

  // fold softmax scale (1/sqrt(64)) and log2(e) into the Q projection
  const float QSCALE = 0.125f * 1.44269504088896340736f;

  dim3 blk(256);
  dim3 gg(D / 128, M / 64);       // (8, 128)

  gemm_bias_kernel<float,  false><<<gg, blk, 0, stream>>>(q, wq, bq, Qb, M, D, D, QSCALE);
  gemm_bias_kernel<float,  false><<<gg, blk, 0, stream>>>(k, wk, bk, Kb, M, D, D, 1.0f);
  gemm_bias_kernel<float,  false><<<gg, blk, 0, stream>>>(v, wv, bv, Vb, M, D, D, 1.0f);

  attn_kernel<<<dim3(S / 128, B * H), blk, 0, stream>>>(Qb, Kb, Vb, Ob, S);

  gemm_bias_kernel<__bf16, true ><<<gg, blk, 0, stream>>>(Ob, wo, bo, d_out, M, D, D, 1.0f);
}
